// MultiHeadAttention_20117626814575
// MI455X (gfx1250) — matrix operand order repaired and checked by reading, not compiled, not run on hardware
//
#include <hip/hip_runtime.h>
#include <hip/hip_bf16.h>

// ---------------- problem constants ----------------
constexpr int Bc  = 2;
constexpr int Sc  = 4096;
constexpr int Hc  = 512;
constexpr int NHc = 8;
constexpr int DHc = 64;   // Hc / NHc

// ---------------- vector types ----------------
typedef __attribute__((ext_vector_type(16))) __bf16          v16bf;
typedef __attribute__((ext_vector_type(8)))  float           v8f;
typedef __attribute__((ext_vector_type(4)))  float           v4f;
typedef __attribute__((ext_vector_type(16))) unsigned short  v16us;
typedef __attribute__((ext_vector_type(8)))  unsigned short  v8us;

__device__ __forceinline__ unsigned short f2bf(float f) {
  unsigned u = __builtin_bit_cast(unsigned, f);
  u += 0x7FFFu + ((u >> 16) & 1u);           // round-to-nearest-even
  return (unsigned short)(u >> 16);
}

__device__ __forceinline__ v8f wmma_bf16(v16bf a, v16bf b, v8f c) {
  return __builtin_amdgcn_wmma_f32_16x16x32_bf16(
      /*neg_a=*/false, a, /*neg_b=*/false, b,
      /*c_mod=*/(short)0, c, /*reuse_a=*/false, /*reuse_b=*/false);
}

// A fragment: 16x32 bf16, row-major source. rowp -> this lane's row at k-step base.
// Lane l: M = l&15; lanes 0-15 hold K = [0..7]+[16..23], lanes 16-31 hold K = [8..15]+[24..31].
__device__ __forceinline__ v16bf make_a(const unsigned short* rowp, int koff) {
  v8us c0 = *(const v8us*)(rowp + koff);
  v8us c1 = *(const v8us*)(rowp + 16 + koff);
  v16us a;
#pragma unroll
  for (int i = 0; i < 8; ++i) { a[i] = c0[i]; a[i + 8] = c1[i]; }
  return __builtin_bit_cast(v16bf, a);
}

// B fragment: 32x16 bf16. Lane l: N = l&15; lanes 0-15 hold K = [0..7]+[16..23], lanes 16-31 hold K = [8..15]+[24..31].
__device__ __forceinline__ v16bf make_b(const unsigned short* p) {
  v8us c0 = *(const v8us*)(p);
  v8us c1 = *(const v8us*)(p + 16);
  v16us b;
#pragma unroll
  for (int i = 0; i < 8; ++i) { b[i] = c0[i]; b[i + 8] = c1[i]; }
  return __builtin_bit_cast(v16bf, b);
}

// A fragment built from f32 LDS row (converted to bf16 on the fly).
__device__ __forceinline__ v16bf make_a_from_f32(const float* rowp, int koff) {
  const v4f* p0 = (const v4f*)(rowp + koff);
  const v4f* p1 = (const v4f*)(rowp + 16 + koff);
  v4f f0 = p0[0], f1 = p0[1], f2 = p1[0], f3 = p1[1];
  v16us a;
#pragma unroll
  for (int e = 0; e < 4; ++e) {
    a[e]      = f2bf(f0[e]);
    a[e + 4]  = f2bf(f1[e]);
    a[e + 8]  = f2bf(f2[e]);
    a[e + 12] = f2bf(f3[e]);
  }
  return __builtin_bit_cast(v16bf, a);
}

// ---------------- stage 0: f32 -> bf16 convert (streaming, non-temporal reads) ----------------
__global__ __launch_bounds__(256) void k_cvt_bf16(const float* __restrict__ src,
                                                  unsigned short* __restrict__ dst,
                                                  int n) {
  int i = (blockIdx.x * 256 + threadIdx.x) * 8;
  if (i < n) {
    const v4f* s4 = (const v4f*)(src + i);
    v4f f0 = __builtin_nontemporal_load(s4);
    v4f f1 = __builtin_nontemporal_load(s4 + 1);
    v8us o;
#pragma unroll
    for (int j = 0; j < 4; ++j) { o[j] = f2bf(f0[j]); o[j + 4] = f2bf(f1[j]); }
    *(v8us*)(dst + i) = o;
  }
}

// ---------------- stage 1: fused Q/K projection (+ bias, + K transpose) ----------------
// One WG = 16 rows x 512 cols of both Q and K. 8 waves, wave w -> cols [64w, 64w+64).
// Ping-pong double buffered; sched_barrier pins loads(k+1) ahead of WMMAs(k).
__global__ __launch_bounds__(256) void k_qk_proj(
    const unsigned short* __restrict__ xb,
    const unsigned short* __restrict__ wqb, const float* __restrict__ bq,
    const unsigned short* __restrict__ wkb, const float* __restrict__ bk,
    unsigned short* __restrict__ Qb, unsigned short* __restrict__ Kb,
    unsigned short* __restrict__ Kt) {
  const int row0 = blockIdx.x * 16;            // [0, B*S)
  const int lane = threadIdx.x & 31;
  const int wave = threadIdx.x >> 5;
  const int colbase = wave * 64;
  const int m    = lane & 15;
  const int koff = (lane >> 4) * 8;
  const int n    = lane & 15;
  const int kb2  = (lane >> 4) * 8;

  v8f accQ[4], accK[4];
#pragma unroll
  for (int t = 0; t < 4; ++t)
#pragma unroll
    for (int r = 0; r < 8; ++r) { accQ[t][r] = 0.f; accK[t][r] = 0.f; }

  const unsigned short* xrow = xb + (size_t)(row0 + m) * Hc;

  v16bf qA[4], kA[4], qB[4], kB[4];
  v16bf aA, aB;
#pragma unroll
  for (int t = 0; t < 4; ++t) {
    const size_t wo = (size_t)(colbase + t * 16 + n) * Hc + kb2;
    qA[t] = make_b(wqb + wo);
    kA[t] = make_b(wkb + wo);
  }
  aA = make_a(xrow, koff);

#pragma unroll 1
  for (int ks = 0; ks < 16; ks += 2) {
    // issue loads for ks+1 while WMMAs of ks run
#pragma unroll
    for (int t = 0; t < 4; ++t) {
      const size_t wo = (size_t)(colbase + t * 16 + n) * Hc + (ks + 1) * 32 + kb2;
      qB[t] = make_b(wqb + wo);
      kB[t] = make_b(wkb + wo);
    }
    aB = make_a(xrow + (ks + 1) * 32, koff);
    __builtin_amdgcn_sched_barrier(0);
#pragma unroll
    for (int t = 0; t < 4; ++t) {
      accQ[t] = wmma_bf16(aA, qA[t], accQ[t]);
      accK[t] = wmma_bf16(aA, kA[t], accK[t]);
    }
    // issue loads for ks+2 (wrap on last) while WMMAs of ks+1 run
    const int ks2 = (ks + 2) & 15;
#pragma unroll
    for (int t = 0; t < 4; ++t) {
      const size_t wo = (size_t)(colbase + t * 16 + n) * Hc + ks2 * 32 + kb2;
      qA[t] = make_b(wqb + wo);
      kA[t] = make_b(wkb + wo);
    }
    aA = make_a(xrow + ks2 * 32, koff);
    __builtin_amdgcn_sched_barrier(0);
#pragma unroll
    for (int t = 0; t < 4; ++t) {
      accQ[t] = wmma_bf16(aB, qB[t], accQ[t]);
      accK[t] = wmma_bf16(aB, kB[t], accK[t]);
    }
  }

  const int bidx = row0 / Sc;                  // batch index (block never spans batches)
#pragma unroll
  for (int t = 0; t < 4; ++t) {
    const int col = colbase + t * 16 + n;
    const float bQv = bq[col], bKv = bk[col];
    const int h = col / DHc, d = col % DHc;
#pragma unroll
    for (int r = 0; r < 8; ++r) {
      const int mm = r + ((lane >> 4) << 3);
      const int grow = row0 + mm;
      unsigned short qv = f2bf(accQ[t][r] + bQv);
      unsigned short kv = f2bf(accK[t][r] + bKv);
      Qb[(size_t)grow * Hc + col] = qv;
      Kb[(size_t)grow * Hc + col] = kv;
      const int s = grow - bidx * Sc;
      Kt[((size_t)(bidx * NHc + h) * DHc + d) * Sc + s] = kv;
    }
  }
}

// ---------------- stage 2: fused scores + softmax + P-write + PV ----------------
// One WG = one (b,h), 16 query rows. LDS: 16 x 4096 f32 scores (256 KB, reused for reduction).
__global__ __launch_bounds__(256) void k_attn(
    const unsigned short* __restrict__ Qb, const unsigned short* __restrict__ Kb,
    const unsigned short* __restrict__ Kt, const float* __restrict__ mask,
    float* __restrict__ scores_out, unsigned short* __restrict__ avb) {
  extern __shared__ float smem[];              // 16*4096 floats

  const int qt = blockIdx.x & (Sc / 16 - 1);   // 256 q-tiles per (b,h)
  const int bh = blockIdx.x >> 8;
  const int b  = bh >> 3;
  const int h  = bh & 7;
  const int q0 = qt * 16;
  const int lane = threadIdx.x & 31;
  const int wave = threadIdx.x >> 5;
  const int m    = lane & 15;
  const int koff = (lane >> 4) * 8;
  const int n    = lane & 15;
  const int kb2  = (lane >> 4) * 8;

  // ---- Phase 1: scores tile (this wave: keys [512w, 512w+512)), ping-pong pipelined ----
  const unsigned short* qrow = Qb + (size_t)(b * Sc + q0 + m) * Hc + h * DHc;
  v16bf aq0 = make_a(qrow, koff);
  v16bf aq1 = make_a(qrow + 32, koff);
  const float* maskb = mask + (size_t)b * Sc;
  const unsigned short* kcol = Kb + ((size_t)b * Sc + n) * Hc + h * DHc + kb2;

  v16bf b0A = make_b(kcol + (size_t)(wave * 512) * Hc);
  v16bf b1A = make_b(kcol + (size_t)(wave * 512) * Hc + 32);
  v16bf b0B, b1B;
#pragma unroll 1
  for (int tt = 0; tt < 32; tt += 2) {
    const int key0 = wave * 512 + tt * 16;
    const int key1 = key0 + 16;
    b0B = make_b(kcol + (size_t)key1 * Hc);
    b1B = make_b(kcol + (size_t)key1 * Hc + 32);
    __builtin_amdgcn_sched_barrier(0);
    {
      v8f c;
#pragma unroll
      for (int r = 0; r < 8; ++r) c[r] = 0.f;
      c = wmma_bf16(aq0, b0A, c);
      c = wmma_bf16(aq1, b1A, c);
      const int col = key0 + n;
      const float mv = maskb[col];
#pragma unroll
      for (int r = 0; r < 8; ++r)
        smem[(r + ((lane >> 4) << 3)) * Sc + col] = c[r] * 0.125f + mv;
    }
    const int key2 = wave * 512 + (((tt + 2) & 31) << 4);
    b0A = make_b(kcol + (size_t)key2 * Hc);
    b1A = make_b(kcol + (size_t)key2 * Hc + 32);
    __builtin_amdgcn_sched_barrier(0);
    {
      v8f c;
#pragma unroll
      for (int r = 0; r < 8; ++r) c[r] = 0.f;
      c = wmma_bf16(aq0, b0B, c);
      c = wmma_bf16(aq1, b1B, c);
      const int col = key1 + n;
      const float mv = maskb[col];
#pragma unroll
      for (int r = 0; r < 8; ++r)
        smem[(r + ((lane >> 4) << 3)) * Sc + col] = c[r] * 0.125f + mv;
    }
  }
  __syncthreads();

  // ---- Phase 2: softmax; wave owns rows 2w and 2w+1; vectorized LDS passes ----
  for (int rr = 0; rr < 2; ++rr) {
    float* row = smem + (wave * 2 + rr) * Sc;
    float mx = -3.0e38f;
    for (int i = lane * 4; i < Sc; i += 128) {
      v4f v = *(const v4f*)(row + i);
      mx = fmaxf(fmaxf(fmaxf(mx, v[0]), fmaxf(v[1], v[2])), v[3]);
    }
#pragma unroll
    for (int off = 16; off; off >>= 1) mx = fmaxf(mx, __shfl_xor(mx, off, 32));
    float sum = 0.f;
    for (int i = lane * 4; i < Sc; i += 128) {
      v4f v = *(const v4f*)(row + i);
#pragma unroll
      for (int j = 0; j < 4; ++j) { v[j] = __expf(v[j] - mx); sum += v[j]; }
      *(v4f*)(row + i) = v;
    }
#pragma unroll
    for (int off = 16; off; off >>= 1) sum += __shfl_xor(sum, off, 32);
    const float inv = 1.0f / sum;
    for (int i = lane * 4; i < Sc; i += 128) {
      v4f v = *(const v4f*)(row + i);
#pragma unroll
      for (int j = 0; j < 4; ++j) v[j] *= inv;
      *(v4f*)(row + i) = v;
    }
  }
  __syncthreads();

  // ---- Phase 3: single streaming NT write of P (only HBM touch of scores; keep L2 for K/Q) ----
  {
    float* dst = scores_out + ((size_t)(b * NHc + h) * Sc + q0) * Sc;
    for (int i = threadIdx.x * 4; i < 16 * Sc; i += 256 * 4)
      __builtin_nontemporal_store(*(const v4f*)(smem + i), (v4f*)(dst + i));
  }

  // ---- Phase 4: PV partials (V = K). A from LDS P (f32->bf16), B from Kt, ping-pong ----
  v8f acc[4];
#pragma unroll
  for (int t = 0; t < 4; ++t)
#pragma unroll
    for (int r = 0; r < 8; ++r) acc[t][r] = 0.f;

  const unsigned short* ktb = Kt + (size_t)(b * NHc + h) * DHc * Sc;  // [d][s]
  const float* prow_base = smem + m * Sc + wave * 512;

  v16bf bfA[4], bfB[4];
  v16bf aA, aB;
#pragma unroll
  for (int t = 0; t < 4; ++t)
    bfA[t] = make_b(ktb + (size_t)(t * 16 + n) * Sc + wave * 512 + kb2);
  aA = make_a_from_f32(prow_base, koff);

#pragma unroll 1
  for (int ks = 0; ks < 16; ks += 2) {
#pragma unroll
    for (int t = 0; t < 4; ++t)
      bfB[t] = make_b(ktb + (size_t)(t * 16 + n) * Sc + wave * 512 + (ks + 1) * 32 + kb2);
    aB = make_a_from_f32(prow_base + (ks + 1) * 32, koff);
    __builtin_amdgcn_sched_barrier(0);
#pragma unroll
    for (int t = 0; t < 4; ++t) acc[t] = wmma_bf16(aA, bfA[t], acc[t]);
    const int ks2 = (ks + 2) & 15;
#pragma unroll
    for (int t = 0; t < 4; ++t)
      bfA[t] = make_b(ktb + (size_t)(t * 16 + n) * Sc + wave * 512 + ks2 * 32 + kb2);
    aA = make_a_from_f32(prow_base + ks2 * 32, koff);
    __builtin_amdgcn_sched_barrier(0);
#pragma unroll
    for (int t = 0; t < 4; ++t) acc[t] = wmma_bf16(aB, bfB[t], acc[t]);
  }
  __syncthreads();                              // all waves done reading P

  // ---- Phase 5: cross-wave reduction in (reused) LDS, write av in bf16 ----
#pragma unroll
  for (int t = 0; t < 4; ++t) {
    const int d = t * 16 + n;
#pragma unroll
    for (int r = 0; r < 8; ++r) {
      const int mm = r + ((lane >> 4) << 3);
      smem[wave * 1024 + mm * 64 + d] = acc[t][r];
    }
  }
  __syncthreads();
#pragma unroll
  for (int j = 0; j < 4; ++j) {
    const int pos = threadIdx.x * 4 + j;        // 1024 positions, 256 threads
    float s = 0.f;
#pragma unroll
    for (int w = 0; w < 8; ++w) s += smem[w * 1024 + pos];
    const int mm = pos >> 6, d = pos & 63;
    avb[(size_t)(b * Sc + q0 + mm) * Hc + h * DHc + d] = f2bf(s);
  }
}

// ---------------- stage 3: out = av @ WO^T + bO (f32 NT output), ping-pong ----------------
__global__ __launch_bounds__(256) void k_out_proj(
    const unsigned short* __restrict__ avb,
    const unsigned short* __restrict__ wob, const float* __restrict__ bo,
    float* __restrict__ out) {
  const int row0 = blockIdx.x * 16;
  const int lane = threadIdx.x & 31;
  const int wave = threadIdx.x >> 5;
  const int colbase = wave * 64;
  const int m    = lane & 15;
  const int koff = (lane >> 4) * 8;
  const int n    = lane & 15;
  const int kb2  = (lane >> 4) * 8;

  v8f acc[4];
#pragma unroll
  for (int t = 0; t < 4; ++t)
#pragma unroll
    for (int r = 0; r < 8; ++r) acc[t][r] = 0.f;

  const unsigned short* arow = avb + (size_t)(row0 + m) * Hc;

  v16bf bfA[4], bfB[4];
  v16bf aA, aB;
#pragma unroll
  for (int t = 0; t < 4; ++t)
    bfA[t] = make_b(wob + (size_t)(colbase + t * 16 + n) * Hc + kb2);
  aA = make_a(arow, koff);

#pragma unroll 1
  for (int ks = 0; ks < 16; ks += 2) {
#pragma unroll
    for (int t = 0; t < 4; ++t)
      bfB[t] = make_b(wob + (size_t)(colbase + t * 16 + n) * Hc + (ks + 1) * 32 + kb2);
    aB = make_a(arow + (ks + 1) * 32, koff);
    __builtin_amdgcn_sched_barrier(0);
#pragma unroll
    for (int t = 0; t < 4; ++t) acc[t] = wmma_bf16(aA, bfA[t], acc[t]);
    const int ks2 = (ks + 2) & 15;
#pragma unroll
    for (int t = 0; t < 4; ++t)
      bfA[t] = make_b(wob + (size_t)(colbase + t * 16 + n) * Hc + ks2 * 32 + kb2);
    aA = make_a(arow + ks2 * 32, koff);
    __builtin_amdgcn_sched_barrier(0);
#pragma unroll
    for (int t = 0; t < 4; ++t) acc[t] = wmma_bf16(aB, bfB[t], acc[t]);
  }

#pragma unroll
  for (int t = 0; t < 4; ++t) {
    const int col = colbase + t * 16 + n;
    const float bv = bo[col];
#pragma unroll
    for (int r = 0; r < 8; ++r) {
      const int mm = r + ((lane >> 4) << 3);
      __builtin_nontemporal_store(acc[t][r] + bv, &out[(size_t)(row0 + mm) * Hc + col]);
    }
  }
}

// ---------------- host launcher ----------------
extern "C" void kernel_launch(void* const* d_in, const int* in_sizes, int n_in,
                              void* d_out, int out_size, void* d_ws, size_t ws_size,
                              hipStream_t stream) {
  (void)in_sizes; (void)n_in; (void)out_size; (void)ws_size;
  const float* x    = (const float*)d_in[0];
  const float* mask = (const float*)d_in[1];
  const float* WQ   = (const float*)d_in[2];
  const float* bQ   = (const float*)d_in[3];
  const float* WK   = (const float*)d_in[4];
  const float* bK   = (const float*)d_in[5];
  const float* WO   = (const float*)d_in[6];
  const float* bO   = (const float*)d_in[7];

  float* out    = (float*)d_out;                       // [B,S,H]
  float* scores = out + (size_t)Bc * Sc * Hc;          // [B,nh,S,S]

  constexpr size_t NX = (size_t)Bc * Sc * Hc;          // 4,194,304
  constexpr size_t NW = (size_t)Hc * Hc;               //   262,144

  unsigned char* w = (unsigned char*)d_ws;
  unsigned short* xb  = (unsigned short*)w;              w += NX * 2;
  unsigned short* wqb = (unsigned short*)w;              w += NW * 2;
  unsigned short* wkb = (unsigned short*)w;              w += NW * 2;
  unsigned short* wob = (unsigned short*)w;              w += NW * 2;
  unsigned short* Qb  = (unsigned short*)w;              w += NX * 2;
  unsigned short* Kb  = (unsigned short*)w;              w += NX * 2;
  unsigned short* Kt  = (unsigned short*)w;              w += NX * 2;
  unsigned short* avb = (unsigned short*)w;              w += NX * 2;

  // stage 0: bf16 conversions (x and the three weight matrices)
  k_cvt_bf16<<<(int)(NX / 8 / 256), 256, 0, stream>>>(x,  xb,  (int)NX);
  k_cvt_bf16<<<(int)(NW / 8 / 256), 256, 0, stream>>>(WQ, wqb, (int)NW);
  k_cvt_bf16<<<(int)(NW / 8 / 256), 256, 0, stream>>>(WK, wkb, (int)NW);
  k_cvt_bf16<<<(int)(NW / 8 / 256), 256, 0, stream>>>(WO, wob, (int)NW);

  // stage 1: Q/K projection (+ transposed K)
  k_qk_proj<<<(Bc * Sc) / 16, 256, 0, stream>>>(xb, wqb, bQ, wkb, bK, Qb, Kb, Kt);

  // stage 2: fused attention (256 KB dynamic LDS per WG; CDNA5 WGP has 320 KB)
  (void)hipFuncSetAttribute((const void*)k_attn,
                            hipFuncAttributeMaxDynamicSharedMemorySize, 16 * Sc * 4);
  k_attn<<<Bc * NHc * (Sc / 16), 256, 16 * Sc * 4, stream>>>(Qb, Kb, Kt, mask, scores, avb);

  // stage 3: output projection
  k_out_proj<<<(Bc * Sc) / 16, 256, 0, stream>>>(avb, wob, bO, out);
}
